// GNNEncoder_20220706030175
// MI455X (gfx1250) — matrix operand order repaired and checked by reading, not compiled, not run on hardware
//
#include <hip/hip_runtime.h>

#define NN    16384
#define NEDGE 262144
#define HD    128
#define NL    3
#define NRED  8192

typedef __attribute__((ext_vector_type(16))) __bf16 v16bf;
typedef __attribute__((ext_vector_type(8)))  float  v8f;

union BfVec {
    v16bf bf;
    uint4 q[2];
};

__device__ __forceinline__ unsigned short f2bf(float f) {
    unsigned int u = __float_as_uint(f);
    u += 0x7FFFu + ((u >> 16) & 1u);            // round-to-nearest-even
    return (unsigned short)(u >> 16);
}
__device__ __forceinline__ unsigned int pack2bf(float a, float b) {
    return (unsigned int)f2bf(a) | ((unsigned int)f2bf(b) << 16);
}
__device__ __forceinline__ float bf2f(unsigned short s) {
    return __uint_as_float(((unsigned int)s) << 16);
}

// A fragment (16x32 bf16, row-major source). Per ISA: lanes 0-15 hold K {0..7,16..23},
// lanes 16-31 hold K {8..15,24..31}; pairs packed low=even-k.
__device__ __forceinline__ v16bf load_a(const unsigned short* row, int hf) {
    BfVec t;
    t.q[0] = *(const uint4*)(row + 8 * hf);
    t.q[1] = *(const uint4*)(row + 16 + 8 * hf);
    return t.bf;
}
// B fragment (32x16 bf16) from pre-swizzled weights: Ws[col][m] = pack(W[2m][col], W[2m+1][col]).
// Lanes 0-15 hold K {0..7,16..23}, lanes 16-31 hold K {8..15,24..31} (matches A).
__device__ __forceinline__ v16bf load_b(const unsigned int* wcol, int hf) {
    BfVec t;
    t.q[0] = *(const uint4*)(wcol + 4 * hf);
    t.q[1] = *(const uint4*)(wcol + 8 + 4 * hf);
    return t.bf;
}
// K-tail fragment: only K=256 (pk lo) and K=257 (pk hi) nonzero -> lanes 0-15, chunk 0, pair 0.
__device__ __forceinline__ v16bf make_a_tail(unsigned int pk) {
    BfVec t;
    t.q[0] = make_uint4(pk, 0u, 0u, 0u);
    t.q[1] = make_uint4(0u, 0u, 0u, 0u);
    return t.bf;
}
__device__ __forceinline__ v8f wmma_bf16(v16bf a, v16bf b, v8f c) {
    return __builtin_amdgcn_wmma_f32_16x16x32_bf16(false, a, false, b, (short)0, c, false, false);
}

// Ws[c*K2 + m] = pack(W[2m][c], W[2m+1][c]); zero pad past Kreal. W row pitch = ncols.
__device__ __forceinline__ void fill_w(unsigned int* Ws, const float* Wg,
                                       int Kreal, int K2, int ncols, int tid, int nthr) {
    for (int i = tid; i < ncols * K2; i += nthr) {
        int c = i / K2, m = i - c * K2, k = 2 * m;
        float f0 = (k     < Kreal) ? Wg[(size_t)k * ncols + c]       : 0.f;
        float f1 = (k + 1 < Kreal) ? Wg[(size_t)(k + 1) * ncols + c] : 0.f;
        Ws[i] = pack2bf(f0, f1);
    }
}

__device__ __forceinline__ void top3_insert(float d, int idx, float* dd, int* ii) {
    if (d < dd[2]) {
        if (d < dd[1]) {
            dd[2] = dd[1]; ii[2] = ii[1];
            if (d < dd[0]) { dd[1] = dd[0]; ii[1] = ii[0]; dd[0] = d; ii[0] = idx; }
            else           { dd[1] = d;     ii[1] = idx; }
        } else { dd[2] = d; ii[2] = idx; }
    }
}

// ---------------- Stage 1: node encoder  u = LN(relu(x@w1+b1)@w2+b2) ----------------
__global__ __launch_bounds__(128) void k_encode(
    const float* __restrict__ x, const float* __restrict__ w1, const float* __restrict__ b1,
    const float* __restrict__ w2g, const float* __restrict__ b2,
    const float* __restrict__ g, const float* __restrict__ bb,
    float* __restrict__ u, unsigned short* __restrict__ ubf)
{
    extern __shared__ unsigned char smem[];
    unsigned int*   Ws = (unsigned int*)smem;                      // 128*64 uints = 32KB
    unsigned short* Hs = (unsigned short*)(smem + 32768);          // 64*128  = 16KB
    float*          Os = (float*)(smem + 32768 + 16384);           // 64*128 f32 = 32KB
    const int tid = threadIdx.x;
    const int tileb = blockIdx.x * 64;

    for (int i = tid; i < 64 * HD; i += 128) {                     // K=3 front layer: VALU
        int r = i >> 7, c = i & 127;
        const float* xr = x + (size_t)(tileb + r) * 3;
        float h = b1[c] + xr[0] * w1[c] + xr[1] * w1[HD + c] + xr[2] * w1[2 * HD + c];
        Hs[i] = f2bf(h > 0.f ? h : 0.f);
    }
    fill_w(Ws, w2g, HD, 64, HD, tid, 128);
    __syncthreads();

    const int lane = tid & 31, wave = tid >> 5, hf = lane >> 4, ln16 = lane & 15;
    const unsigned short* hrow = Hs + (size_t)(wave * 16 + ln16) * HD;
    for (int nt = 0; nt < 8; ++nt) {
        v8f acc = {};
        const unsigned int* wcol = Ws + (size_t)(nt * 16 + ln16) * 64;
        for (int s = 0; s < 4; ++s)
            acc = wmma_bf16(load_a(hrow + s * 32, hf), load_b(wcol + s * 16, hf), acc);
        int colb = nt * 16 + ln16;
        float bias = b2[colb];
        int rb = wave * 16 + hf * 8;
#pragma unroll
        for (int r = 0; r < 8; ++r)
            Os[(rb + r) * HD + colb] = acc[r] + bias;
    }
    __syncthreads();
    if (tid < 64) {
        int node = tileb + tid;
        const float* o = Os + tid * HD;
        float m = 0.f;
        for (int c = 0; c < HD; ++c) m += o[c];
        m *= (1.f / HD);
        float var = 0.f;
        for (int c = 0; c < HD; ++c) { float d = o[c] - m; var += d * d; }
        float inv = rsqrtf(var * (1.f / HD) + 1e-5f);
        for (int c = 0; c < HD; ++c) {
            float val = (o[c] - m) * inv * g[c] + bb[c];
            u[(size_t)node * HD + c] = val;
            ubf[(size_t)node * HD + c] = f2bf(val);
        }
    }
}

// ---------------- Stage 2: edge features (ED=2 MLP, VALU) ----------------
__global__ __launch_bounds__(256) void k_edge_init(
    const float* __restrict__ pos, const int* __restrict__ rowi, const int* __restrict__ coli,
    const float* __restrict__ w1, const float* __restrict__ b1,
    const float* __restrict__ w2, const float* __restrict__ b2,
    const float* __restrict__ g, const float* __restrict__ bb,
    float* __restrict__ e)
{
    int eid = blockIdx.x * 256 + threadIdx.x;
    int r = rowi[eid], c = coli[eid];
    float e0 = pos[(size_t)c * 2]     - pos[(size_t)r * 2];
    float e1 = pos[(size_t)c * 2 + 1] - pos[(size_t)r * 2 + 1];
    float o0 = b2[0], o1 = b2[1];
    for (int j = 0; j < HD; ++j) {
        float h = b1[j] + e0 * w1[j] + e1 * w1[HD + j];
        h = fmaxf(h, 0.f);
        o0 += h * w2[j * 2]; o1 += h * w2[j * 2 + 1];
    }
    float m = 0.5f * (o0 + o1);
    float var = 0.5f * ((o0 - m) * (o0 - m) + (o1 - m) * (o1 - m));
    float inv = rsqrtf(var + 1e-5f);
    e[(size_t)eid * 2]     = (o0 - m) * inv * g[0] + bb[0];
    e[(size_t)eid * 2 + 1] = (o1 - m) * inv * g[1] + bb[1];
}

__global__ void k_count(const int* __restrict__ coli, float* __restrict__ cnt) {
    int e = blockIdx.x * 256 + threadIdx.x;
    atomicAdd(&cnt[coli[e]], 1.0f);
}

// ---------------- Stage 3a: message MLP + scatter-sum (WMMA, gathered A) ----------------
__global__ __launch_bounds__(256) void k_msg(
    const unsigned short* __restrict__ ubf, const float* __restrict__ e,
    const int* __restrict__ rowi, const int* __restrict__ coli,
    const float* __restrict__ w1g, const float* __restrict__ b1,
    const float* __restrict__ w2g, const float* __restrict__ b2,
    float* __restrict__ agg)
{
    extern __shared__ unsigned char smem[];
    unsigned int*   Ws = (unsigned int*)smem;                      // 128 cols * 144 uints = 72KB
    unsigned short* Hs = (unsigned short*)(smem + 73728);          // 128*128 = 32KB
    const int tid = threadIdx.x, tileb = blockIdx.x * 128;
    fill_w(Ws, w1g, 258, 144, HD, tid, 256);
    __syncthreads();

    const int lane = tid & 31, wave = tid >> 5, hf = lane >> 4, ln16 = lane & 15;
    const int eid = tileb + wave * 16 + ln16;
    const int nr = rowi[eid], nc = coli[eid];
    const unsigned short* ar0 = ubf + (size_t)nr * HD;
    const unsigned short* ar1 = ubf + (size_t)nc * HD;
    unsigned int epk = (hf == 0) ? pack2bf(e[(size_t)eid * 2], e[(size_t)eid * 2 + 1]) : 0u;

    for (int nt = 0; nt < 8; ++nt) {
        v8f acc = {};
        const unsigned int* wcol = Ws + (size_t)(nt * 16 + ln16) * 144;
        for (int s = 0; s < 9; ++s) {
            v16bf a;
            if (s < 4)      a = load_a(ar0 + s * 32, hf);
            else if (s < 8) a = load_a(ar1 + (s - 4) * 32, hf);
            else            a = make_a_tail(epk);
            acc = wmma_bf16(a, load_b(wcol + s * 16, hf), acc);
        }
        int colb = nt * 16 + ln16;
        float bias = b1[colb];
        int rb = wave * 16 + hf * 8;
#pragma unroll
        for (int r = 0; r < 8; ++r) {
            float v = acc[r] + bias;
            Hs[(rb + r) * HD + colb] = f2bf(v > 0.f ? v : 0.f);
        }
    }
    __syncthreads();
    fill_w(Ws, w2g, HD, 64, HD, tid, 256);                         // reuse Ws region for W2
    __syncthreads();

    const unsigned short* hrow = Hs + (size_t)(wave * 16 + ln16) * HD;
    for (int nt = 0; nt < 8; ++nt) {
        v8f acc = {};
        const unsigned int* wcol = Ws + (size_t)(nt * 16 + ln16) * 64;
        for (int s = 0; s < 4; ++s)
            acc = wmma_bf16(load_a(hrow + s * 32, hf), load_b(wcol + s * 16, hf), acc);
        int colb = nt * 16 + ln16;
        float bias = b2[colb];
        int rb = wave * 16 + hf * 8;
#pragma unroll
        for (int r = 0; r < 8; ++r) {
            int ed = tileb + rb + r;
            atomicAdd(&agg[(size_t)coli[ed] * HD + colb], acc[r] + bias);
        }
    }
}

__global__ void k_aggdiv(const float* __restrict__ agg, const float* __restrict__ cnt,
                         unsigned short* __restrict__ aggb) {
    int i = blockIdx.x * 256 + threadIdx.x;
    float c = cnt[i >> 7];
    c = c > 1.f ? c : 1.f;
    aggb[i] = f2bf(agg[i] / c);
}

// ---------------- Stage 3b: node update u = LN(u + MLP([u, agg])) ----------------
__global__ __launch_bounds__(128) void k_node(
    const unsigned short* __restrict__ aggb,
    const float* __restrict__ w1g, const float* __restrict__ b1,
    const float* __restrict__ w2g, const float* __restrict__ b2,
    const float* __restrict__ g, const float* __restrict__ bb,
    float* __restrict__ u, unsigned short* __restrict__ ubf)
{
    extern __shared__ unsigned char smem[];
    unsigned int*   Ws = (unsigned int*)smem;                      // 128*128 uints = 64KB (K=256)
    unsigned short* Hs = (unsigned short*)(smem + 65536);          // 16KB
    float*          Os = (float*)(smem + 65536 + 16384);           // 32KB
    const int tid = threadIdx.x, tileb = blockIdx.x * 64;
    fill_w(Ws, w1g, 256, 128, HD, tid, 128);
    __syncthreads();

    const int lane = tid & 31, wave = tid >> 5, hf = lane >> 4, ln16 = lane & 15;
    const int node = tileb + wave * 16 + ln16;
    const unsigned short* ar0 = ubf  + (size_t)node * HD;
    const unsigned short* ar1 = aggb + (size_t)node * HD;
    for (int nt = 0; nt < 8; ++nt) {
        v8f acc = {};
        const unsigned int* wcol = Ws + (size_t)(nt * 16 + ln16) * 128;
        for (int s = 0; s < 8; ++s) {
            v16bf a;
            if (s < 4) a = load_a(ar0 + s * 32, hf);
            else       a = load_a(ar1 + (s - 4) * 32, hf);
            acc = wmma_bf16(a, load_b(wcol + s * 16, hf), acc);
        }
        int colb = nt * 16 + ln16;
        float bias = b1[colb];
        int rb = wave * 16 + hf * 8;
#pragma unroll
        for (int r = 0; r < 8; ++r) {
            float v = acc[r] + bias;
            Hs[(rb + r) * HD + colb] = f2bf(v > 0.f ? v : 0.f);
        }
    }
    __syncthreads();
    fill_w(Ws, w2g, HD, 64, HD, tid, 128);
    __syncthreads();
    const unsigned short* hrow = Hs + (size_t)(wave * 16 + ln16) * HD;
    for (int nt = 0; nt < 8; ++nt) {
        v8f acc = {};
        const unsigned int* wcol = Ws + (size_t)(nt * 16 + ln16) * 64;
        for (int s = 0; s < 4; ++s)
            acc = wmma_bf16(load_a(hrow + s * 32, hf), load_b(wcol + s * 16, hf), acc);
        int colb = nt * 16 + ln16;
        float bias = b2[colb];
        int rb = wave * 16 + hf * 8;
#pragma unroll
        for (int r = 0; r < 8; ++r)
            Os[(rb + r) * HD + colb] = acc[r] + bias;
    }
    __syncthreads();
    if (tid < 64) {
        int nd = tileb + tid;
        float* o = Os + tid * HD;
        float m = 0.f;
        for (int c = 0; c < HD; ++c) { float v = o[c] + u[(size_t)nd * HD + c]; o[c] = v; m += v; }
        m *= (1.f / HD);
        float var = 0.f;
        for (int c = 0; c < HD; ++c) { float d = o[c] - m; var += d * d; }
        float inv = rsqrtf(var * (1.f / HD) + 1e-5f);
        for (int c = 0; c < HD; ++c) {
            float val = (o[c] - m) * inv * g[c] + bb[c];
            u[(size_t)nd * HD + c] = val;
            ubf[(size_t)nd * HD + c] = f2bf(val);
        }
    }
}

// ---------------- Stage 3c: edge update e = LN(e + MLP([u_r, u_c, e])) ----------------
__global__ __launch_bounds__(256) void k_edgeup(
    const unsigned short* __restrict__ ubf,
    const int* __restrict__ rowi, const int* __restrict__ coli,
    const float* __restrict__ w1g, const float* __restrict__ b1,
    const float* __restrict__ w2g, const float* __restrict__ b2,
    const float* __restrict__ g, const float* __restrict__ bb,
    float* __restrict__ e)
{
    extern __shared__ unsigned char smem[];
    unsigned int*   Ws = (unsigned int*)smem;                      // 72KB
    unsigned short* Hs = (unsigned short*)(smem + 73728);          // 32KB
    const int tid = threadIdx.x, tileb = blockIdx.x * 128;
    fill_w(Ws, w1g, 258, 144, HD, tid, 256);
    __syncthreads();

    const int lane = tid & 31, wave = tid >> 5, hf = lane >> 4, ln16 = lane & 15;
    const int eid = tileb + wave * 16 + ln16;
    const int nr = rowi[eid], nc = coli[eid];
    const unsigned short* ar0 = ubf + (size_t)nr * HD;
    const unsigned short* ar1 = ubf + (size_t)nc * HD;
    unsigned int epk = (hf == 0) ? pack2bf(e[(size_t)eid * 2], e[(size_t)eid * 2 + 1]) : 0u;

    for (int nt = 0; nt < 8; ++nt) {
        v8f acc = {};
        const unsigned int* wcol = Ws + (size_t)(nt * 16 + ln16) * 144;
        for (int s = 0; s < 9; ++s) {
            v16bf a;
            if (s < 4)      a = load_a(ar0 + s * 32, hf);
            else if (s < 8) a = load_a(ar1 + (s - 4) * 32, hf);
            else            a = make_a_tail(epk);
            acc = wmma_bf16(a, load_b(wcol + s * 16, hf), acc);
        }
        int colb = nt * 16 + ln16;
        float bias = b1[colb];
        int rb = wave * 16 + hf * 8;
#pragma unroll
        for (int r = 0; r < 8; ++r) {
            float v = acc[r] + bias;
            Hs[(rb + r) * HD + colb] = f2bf(v > 0.f ? v : 0.f);
        }
    }
    __syncthreads();
    if (tid < 128) {                                               // 128->2 head: VALU
        int ed = tileb + tid;
        const unsigned short* h = Hs + (size_t)tid * HD;
        float o0 = b2[0], o1 = b2[1];
        for (int j = 0; j < HD; ++j) {
            float hv = bf2f(h[j]);
            o0 += hv * w2g[j * 2]; o1 += hv * w2g[j * 2 + 1];
        }
        float v0 = e[(size_t)ed * 2] + o0, v1 = e[(size_t)ed * 2 + 1] + o1;
        float m = 0.5f * (v0 + v1);
        float var = 0.5f * ((v0 - m) * (v0 - m) + (v1 - m) * (v1 - m));
        float inv = rsqrtf(var + 1e-5f);
        e[(size_t)ed * 2]     = (v0 - m) * inv * g[0] + bb[0];
        e[(size_t)ed * 2 + 1] = (v1 - m) * inv * g[1] + bb[1];
    }
}

// ---------------- Stage 4: u3 = LN(MLP(u)) ----------------
__global__ __launch_bounds__(128) void k_proj(
    const unsigned short* __restrict__ ubf,
    const float* __restrict__ w1g, const float* __restrict__ b1,
    const float* __restrict__ w2g, const float* __restrict__ b2,
    const float* __restrict__ g, const float* __restrict__ bb,
    float* __restrict__ u3)
{
    extern __shared__ unsigned char smem[];
    unsigned int*   Ws = (unsigned int*)smem;                      // 32KB
    unsigned short* Hs = (unsigned short*)(smem + 32768);          // 16KB
    float*          Os = (float*)(smem + 32768 + 16384);           // 32KB
    const int tid = threadIdx.x, tileb = blockIdx.x * 64;
    fill_w(Ws, w1g, HD, 64, HD, tid, 128);
    __syncthreads();

    const int lane = tid & 31, wave = tid >> 5, hf = lane >> 4, ln16 = lane & 15;
    const int node = tileb + wave * 16 + ln16;
    const unsigned short* ar = ubf + (size_t)node * HD;
    for (int nt = 0; nt < 8; ++nt) {
        v8f acc = {};
        const unsigned int* wcol = Ws + (size_t)(nt * 16 + ln16) * 64;
        for (int s = 0; s < 4; ++s)
            acc = wmma_bf16(load_a(ar + s * 32, hf), load_b(wcol + s * 16, hf), acc);
        int colb = nt * 16 + ln16;
        float bias = b1[colb];
        int rb = wave * 16 + hf * 8;
#pragma unroll
        for (int r = 0; r < 8; ++r) {
            float v = acc[r] + bias;
            Hs[(rb + r) * HD + colb] = f2bf(v > 0.f ? v : 0.f);
        }
    }
    __syncthreads();
    fill_w(Ws, w2g, HD, 64, HD, tid, 128);
    __syncthreads();
    const unsigned short* hrow = Hs + (size_t)(wave * 16 + ln16) * HD;
    for (int nt = 0; nt < 8; ++nt) {
        v8f acc = {};
        const unsigned int* wcol = Ws + (size_t)(nt * 16 + ln16) * 64;
        for (int s = 0; s < 4; ++s)
            acc = wmma_bf16(load_a(hrow + s * 32, hf), load_b(wcol + s * 16, hf), acc);
        int colb = nt * 16 + ln16;
        float bias = b2[colb];
        int rb = wave * 16 + hf * 8;
#pragma unroll
        for (int r = 0; r < 8; ++r)
            Os[(rb + r) * HD + colb] = acc[r] + bias;
    }
    __syncthreads();
    if (tid < 64) {
        int nd = tileb + tid;
        const float* o = Os + tid * HD;
        float m = 0.f;
        for (int c = 0; c < HD; ++c) m += o[c];
        m *= (1.f / HD);
        float var = 0.f;
        for (int c = 0; c < HD; ++c) { float d = o[c] - m; var += d * d; }
        float inv = rsqrtf(var * (1.f / HD) + 1e-5f);
        for (int c = 0; c < HD; ++c)
            u3[(size_t)nd * HD + c] = (o[c] - m) * inv * g[c] + bb[c];
    }
}

// ---------------- Stage 5: kNN(K=3) interpolation + mean-pool accumulate ----------------
__global__ __launch_bounds__(256) void k_knn(const float* __restrict__ pos,
                                             const float* __restrict__ u3,
                                             float* __restrict__ zsum)
{
    __shared__ float sd[256 * 3];
    __shared__ int   si[256 * 3];
    __shared__ float sw[3];
    __shared__ int   sk[3];
    const int r = blockIdx.x;
    const int ir = (int)((double)r * (double)(NN - 1) / (double)(NRED - 1));
    const float px = pos[(size_t)ir * 2], py = pos[(size_t)ir * 2 + 1];
    float dd[3] = {3.4e38f, 3.4e38f, 3.4e38f};
    int ii[3] = {0, 0, 0};
    for (int n = threadIdx.x; n < NN; n += 256) {
        float dx = pos[(size_t)n * 2] - px, dy = pos[(size_t)n * 2 + 1] - py;
        top3_insert(dx * dx + dy * dy, n, dd, ii);
    }
    for (int k = 0; k < 3; ++k) { sd[threadIdx.x * 3 + k] = dd[k]; si[threadIdx.x * 3 + k] = ii[k]; }
    __syncthreads();
    if (threadIdx.x == 0) {
        float bd[3] = {3.4e38f, 3.4e38f, 3.4e38f}; int bi[3] = {0, 0, 0};
        for (int t = 0; t < 256 * 3; ++t) top3_insert(sd[t], si[t], bd, bi);
        float w[3], wsum = 0.f;
        for (int k = 0; k < 3; ++k) { w[k] = 1.f / (sqrtf(fmaxf(bd[k], 0.f)) + 1e-8f); wsum += w[k]; }
        for (int k = 0; k < 3; ++k) { sw[k] = w[k] / wsum; sk[k] = bi[k]; }
    }
    __syncthreads();
    if (threadIdx.x < HD) {
        int c = threadIdx.x;
        float acc = sw[0] * u3[(size_t)sk[0] * HD + c]
                  + sw[1] * u3[(size_t)sk[1] * HD + c]
                  + sw[2] * u3[(size_t)sk[2] * HD + c];
        atomicAdd(&zsum[c], acc);
    }
}

// ---------------- Stage 6: mean + output projection ----------------
__global__ __launch_bounds__(128) void k_head(const float* __restrict__ zsum,
                                              const float* __restrict__ w1, const float* __restrict__ b1,
                                              const float* __restrict__ w2, const float* __restrict__ b2,
                                              float* __restrict__ out)
{
    __shared__ float z[HD];
    __shared__ float h[HD];
    int t = threadIdx.x;
    z[t] = zsum[t] * (1.0f / NRED);
    __syncthreads();
    float acc = b1[t];
    for (int i = 0; i < HD; ++i) acc += z[i] * w1[i * HD + t];
    h[t] = fmaxf(acc, 0.f);
    __syncthreads();
    if (t < 64) {
        float o = b2[t];
        for (int j = 0; j < HD; ++j) o += h[j] * w2[j * 64 + t];
        out[t] = o;
    }
}

extern "C" void kernel_launch(void* const* d_in, const int* in_sizes, int n_in,
                              void* d_out, int out_size, void* d_ws, size_t ws_size,
                              hipStream_t stream) {
    (void)in_sizes; (void)n_in; (void)out_size; (void)ws_size;
    const float* x       = (const float*)d_in[0];
    const float* pos     = (const float*)d_in[1];
    const int*   ei      = (const int*)d_in[2];
    const int*   rowi    = ei;
    const int*   coli    = ei + NEDGE;
    const float* m0_w1   = (const float*)d_in[3];
    const float* m0_b1   = (const float*)d_in[4];
    const float* m0_w2   = (const float*)d_in[5];
    const float* m0_b2   = (const float*)d_in[6];
    const float* ln0_g   = (const float*)d_in[7];
    const float* ln0_b   = (const float*)d_in[8];
    const float* m1_w1   = (const float*)d_in[9];
    const float* m1_b1   = (const float*)d_in[10];
    const float* m1_w2   = (const float*)d_in[11];
    const float* m1_b2   = (const float*)d_in[12];
    const float* ln1_g   = (const float*)d_in[13];
    const float* ln1_b   = (const float*)d_in[14];
    const float* gm_w1   = (const float*)d_in[15];
    const float* gm_b1   = (const float*)d_in[16];
    const float* gm_w2   = (const float*)d_in[17];
    const float* gm_b2   = (const float*)d_in[18];
    const float* gn_w1   = (const float*)d_in[19];
    const float* gn_b1   = (const float*)d_in[20];
    const float* gn_w2   = (const float*)d_in[21];
    const float* gn_b2   = (const float*)d_in[22];
    const float* ge_w1   = (const float*)d_in[23];
    const float* ge_b1   = (const float*)d_in[24];
    const float* ge_w2   = (const float*)d_in[25];
    const float* ge_b2   = (const float*)d_in[26];
    const float* lnn_g   = (const float*)d_in[27];
    const float* lnn_b   = (const float*)d_in[28];
    const float* lne_g   = (const float*)d_in[29];
    const float* lne_b   = (const float*)d_in[30];
    const float* m2_w1   = (const float*)d_in[31];
    const float* m2_b1   = (const float*)d_in[32];
    const float* m2_w2   = (const float*)d_in[33];
    const float* m2_b2   = (const float*)d_in[34];
    const float* ln2_g   = (const float*)d_in[35];
    const float* ln2_b   = (const float*)d_in[36];
    const float* op_w1   = (const float*)d_in[37];
    const float* op_b1   = (const float*)d_in[38];
    const float* op_w2   = (const float*)d_in[39];
    const float* op_b2   = (const float*)d_in[40];

    char* ws = (char*)d_ws;
    float*          u    = (float*)(ws + 0);                  // 8 MB
    unsigned short* ubf  = (unsigned short*)(ws + 8388608);   // 4 MB
    float*          u3   = (float*)(ws + 12582912);           // 8 MB
    float*          ef   = (float*)(ws + 20971520);           // 2 MB
    float*          agg  = (float*)(ws + 23068672);           // 8 MB
    unsigned short* aggb = (unsigned short*)(ws + 31457280);  // 4 MB
    float*          cnt  = (float*)(ws + 35651584);           // 64 KB
    float*          zsum = (float*)(ws + 35717120);           // 512 B

    hipMemsetAsync(cnt, 0, (size_t)NN * 4, stream);
    hipMemsetAsync(zsum, 0, HD * 4, stream);

    k_encode<<<NN / 64, 128, 81920, stream>>>(x, m0_w1, m0_b1, m0_w2, m0_b2, ln0_g, ln0_b, u, ubf);
    k_edge_init<<<NEDGE / 256, 256, 0, stream>>>(pos, rowi, coli, m1_w1, m1_b1, m1_w2, m1_b2,
                                                 ln1_g, ln1_b, ef);
    k_count<<<NEDGE / 256, 256, 0, stream>>>(coli, cnt);

    for (int l = 0; l < NL; ++l) {
        hipMemsetAsync(agg, 0, (size_t)NN * HD * 4, stream);
        k_msg<<<NEDGE / 128, 256, 106496, stream>>>(ubf, ef, rowi, coli,
            gm_w1 + (size_t)l * 258 * HD, gm_b1 + (size_t)l * HD,
            gm_w2 + (size_t)l * HD * HD,  gm_b2 + (size_t)l * HD, agg);
        k_aggdiv<<<(NN * HD) / 256, 256, 0, stream>>>(agg, cnt, aggb);
        k_node<<<NN / 64, 128, 114688, stream>>>(aggb,
            gn_w1 + (size_t)l * 256 * HD, gn_b1 + (size_t)l * HD,
            gn_w2 + (size_t)l * HD * HD,  gn_b2 + (size_t)l * HD,
            lnn_g + (size_t)l * HD, lnn_b + (size_t)l * HD, u, ubf);
        k_edgeup<<<NEDGE / 128, 256, 106496, stream>>>(ubf, rowi, coli,
            ge_w1 + (size_t)l * 258 * HD, ge_b1 + (size_t)l * HD,
            ge_w2 + (size_t)l * HD * 2,   ge_b2 + (size_t)l * 2,
            lne_g + (size_t)l * 2, lne_b + (size_t)l * 2, ef);
    }

    k_proj<<<NN / 64, 128, 81920, stream>>>(ubf, m2_w1, m2_b1, m2_w2, m2_b2, ln2_g, ln2_b, u3);
    k_knn<<<NRED, 256, 0, stream>>>(pos, u3, zsum);
    k_head<<<1, 128, 0, stream>>>(zsum, op_w1, op_b1, op_w2, op_b2, (float*)d_out);
}
